// GemNetT_72103910966016
// MI455X (gfx1250) — matrix operand order repaired and checked by reading, not compiled, not run on hardware
//
#include <hip/hip_runtime.h>

using u16 = unsigned short;

#define NATOM   2000
#define NEDGE   40000
#define NTRIP   500000
#define FLAG_SILU 1
#define INV_SQRT2F 0.70710678118654752440f

typedef __attribute__((ext_vector_type(8)))  float  v8f;
typedef __attribute__((ext_vector_type(8)))  __bf16 bf16x8;
typedef __attribute__((ext_vector_type(16))) __bf16 v16bf;

__device__ __forceinline__ u16 f2bf(float f) {
  unsigned u = __float_as_uint(f);
  u += 0x7fffu + ((u >> 16) & 1u);
  return (u16)(u >> 16);
}
__device__ __forceinline__ float siluf(float x) { return x / (1.0f + __expf(-x)); }

// ---------------------------------------------------------------------------
// Generic bf16 WMMA GEMM: C[M,N] = epilogue(A[M,K] @ B[K,N])
// Block: 256 threads (8 wave32 waves), block tile 128x128, BK=32.
// Wave tile 32x64 -> 8 v_wmma_f32_16x16x32_bf16 per K-step per wave.
// A tile staged via GLOBAL_LOAD_ASYNC_TO_LDS_B128 on full tiles (uniform path),
// predicated uint4 loads on edge tiles. All A operands have lda==K, K%8==0,
// so chunk-granular bounds checks are exact (no scalar tail).
// Epilogue: (+bias) -> (silu) -> (*mul) -> ((+add)*addscale) -> store f32/bf16
// ---------------------------------------------------------------------------
__launch_bounds__(256)
__global__ void wmma_gemm_kernel(const u16* __restrict__ A, int lda,
                                 const u16* __restrict__ B, int ldb,
                                 int M, int N, int K,
                                 const float* __restrict__ bias,
                                 const float* __restrict__ mulp,
                                 const float* __restrict__ addp, float addscale,
                                 float* __restrict__ outF,
                                 u16* __restrict__ outB,
                                 int ldc, int flags)
{
  __shared__ __align__(64) u16 As[128][32];   // [m][k]
  __shared__ __align__(64) u16 Bs[128][32];   // [n][k]  (transposed stage)

  const int tid   = threadIdx.x;
  const int lane  = tid & 31;
  const int wave  = tid >> 5;
  const int wm    = wave & 3;    // M sub-tile (x32)
  const int wn    = wave >> 2;   // N sub-tile (x64)
  const int l15   = lane & 15;
  const int khalf = lane >> 4;
  const int row0  = blockIdx.y * 128;
  const int col0  = blockIdx.x * 128;

  const unsigned ldsA = (unsigned)(uintptr_t)&As[0][0];  // LDS byte offset of As
  const bool rows_full = (row0 + 128 <= M);

  v8f acc[2][4];
  {
    v8f zero = {};
    #pragma unroll
    for (int mt = 0; mt < 2; ++mt)
      #pragma unroll
      for (int nt = 0; nt < 4; ++nt) acc[mt][nt] = zero;
  }

  for (int k0 = 0; k0 < K; k0 += 32) {
    const bool full = rows_full && (k0 + 32 <= K);
    if (full) {
      // ---- async stage A tile: per-lane 16B chunk -> contiguous LDS ----
      #pragma unroll
      for (int i = 0; i < 2; ++i) {
        int c = tid + i * 256;                         // 0..511 chunks of 8 u16
        const u16* gp = A + (size_t)(row0 + (c >> 2)) * lda + (k0 + ((c & 3) << 3));
        unsigned laddr = ldsA + (unsigned)c * 16u;
        asm volatile("global_load_async_to_lds_b128 %0, %1, off"
                     :: "v"(laddr), "v"(gp) : "memory");
      }
    } else {
      // ---- predicated vector stage (edge tiles) ----
      #pragma unroll
      for (int i = 0; i < 2; ++i) {
        int c  = tid + i * 256;
        int r  = c >> 2;
        int kc = (c & 3) << 3;                         // 0,8,16,24
        int gr = row0 + r;
        int gk = k0 + kc;
        uint4 val; val.x = val.y = val.z = val.w = 0u;
        if (gr < M && gk + 8 <= K)                     // exact: lda==K, K%8==0
          val = *(const uint4*)(A + (size_t)gr * lda + gk);
        *(uint4*)&As[r][kc] = val;
      }
    }
    // ---- stage B tile transposed into Bs[n][k] (zero-fills K/N tails) ----
    #pragma unroll
    for (int i = 0; i < 16; ++i) {
      int c2 = tid + i * 256;                          // 0..4095
      int kk = c2 >> 7;                                // 0..31
      int nn = c2 & 127;                               // 0..127
      int gk = k0 + kk, gc = col0 + nn;
      u16 v = 0;
      if (gk < K && gc < N) v = B[(size_t)gk * ldb + gc];
      Bs[nn][kk] = v;
    }
    if (k0 + 32 < K)
      __builtin_prefetch(A + (size_t)(row0 + (tid & 127)) * lda + k0 + 32, 0, 0);
    if (full)
      asm volatile("s_wait_asynccnt 0x0" ::: "memory");
    __syncthreads();

    // ---- fragments per ISA 16-bit layouts ----
    const int ar0 = wm * 32 + l15;
    bf16x8 a0lo = *(const bf16x8*)&As[ar0     ][khalf * 8];
    bf16x8 a0hi = *(const bf16x8*)&As[ar0     ][16 + khalf * 8];
    bf16x8 a1lo = *(const bf16x8*)&As[ar0 + 16][khalf * 8];
    bf16x8 a1hi = *(const bf16x8*)&As[ar0 + 16][16 + khalf * 8];
    v16bf afrag[2];
    afrag[0] = __builtin_shufflevector(a0lo, a0hi, 0,1,2,3,4,5,6,7,8,9,10,11,12,13,14,15);
    afrag[1] = __builtin_shufflevector(a1lo, a1hi, 0,1,2,3,4,5,6,7,8,9,10,11,12,13,14,15);

    v16bf bfrag[4];
    #pragma unroll
    for (int nt = 0; nt < 4; ++nt)
    {
      bf16x8 blo = *(const bf16x8*)&Bs[wn * 64 + nt * 16 + l15][khalf * 8];
      bf16x8 bhi = *(const bf16x8*)&Bs[wn * 64 + nt * 16 + l15][16 + khalf * 8];
      bfrag[nt] = __builtin_shufflevector(blo, bhi, 0,1,2,3,4,5,6,7,8,9,10,11,12,13,14,15);
    }

    #pragma unroll
    for (int mt = 0; mt < 2; ++mt)
      #pragma unroll
      for (int nt = 0; nt < 4; ++nt)
        acc[mt][nt] = __builtin_amdgcn_wmma_f32_16x16x32_bf16(
            false, afrag[mt], false, bfrag[nt], (short)0, acc[mt][nt], false, false);
    __syncthreads();
  }

  // ---- epilogue ----
  #pragma unroll
  for (int mt = 0; mt < 2; ++mt) {
    #pragma unroll
    for (int nt = 0; nt < 4; ++nt) {
      int c = col0 + wn * 64 + nt * 16 + l15;
      #pragma unroll
      for (int v = 0; v < 8; ++v) {
        int r = row0 + wm * 32 + mt * 16 + khalf * 8 + v;
        if (r < M && c < N) {
          float x = acc[mt][nt][v];
          if (bias) x += bias[c];
          if (flags & FLAG_SILU) x = siluf(x);
          size_t o = (size_t)r * ldc + c;
          if (mulp) x *= mulp[o];
          if (addp) x = (x + addp[o]) * addscale;
          if (outF) outF[o] = x;
          if (outB) outB[o] = f2bf(x);
        }
      }
    }
  }
}

// ---------------------------------------------------------------------------
// Small support kernels
// ---------------------------------------------------------------------------
__global__ void cvt_kernel(const float* __restrict__ s, u16* __restrict__ d, long long n) {
  long long i = (long long)blockIdx.x * blockDim.x + threadIdx.x;
  if (i < n) d[i] = f2bf(s[i]);
}

__global__ void pack_cbf3_kernel(const float* __restrict__ s, u16* __restrict__ d) {
  int i = blockIdx.x * blockDim.x + threadIdx.x;   // over 7*128*16
  if (i >= 7 * 128 * 16) return;
  int ss = i / (128 * 16);
  int r  = (i / 16) % 128;
  int k  = i & 15;
  d[r * 112 + ss * 16 + k] = f2bf(s[i]);
}

__global__ void rbf_kernel(const float* __restrict__ D, float* __restrict__ of, u16* __restrict__ ob) {
  int i = blockIdx.x * blockDim.x + threadIdx.x;
  if (i >= NEDGE * 128) return;
  int e = i >> 7, r = i & 127;
  float ds   = D[e] * (1.0f / 6.0f);
  float off  = (float)r * (1.0f / 127.0f);
  float diff = ds - off;
  float val  = __expf(-0.5f * 127.0f * 127.0f * diff * diff);
  float d2 = ds * ds;
  float d5 = d2 * d2 * ds;
  float env = 1.0f + d5 * (-21.0f + ds * (35.0f - 15.0f * ds));
  env = (ds < 1.0f) ? env : 0.0f;
  float v = env * val;
  of[i] = v;
  ob[i] = f2bf(v);
}

__global__ void sph_kernel(const float* __restrict__ V, const int* __restrict__ ca,
                           const int* __restrict__ ba, float* __restrict__ sph) {
  int t = blockIdx.x * blockDim.x + threadIdx.x;
  if (t >= NTRIP) return;
  int ea = ca[t] * 3, eb = ba[t] * 3;
  float x = V[ea] * V[eb] + V[ea + 1] * V[eb + 1] + V[ea + 2] * V[eb + 2];
  x = fminf(1.0f, fmaxf(-1.0f, x));
  float Pm2 = 1.0f, Pm1 = x;
  sph[t * 7 + 0] = 0.28209479177f;
  sph[t * 7 + 1] = 0.48860251190f * x;
  #pragma unroll
  for (int l = 2; l < 7; ++l) {
    float P = ((2.0f * l - 1.0f) * x * Pm1 - (l - 1.0f) * Pm2) / (float)l;
    sph[t * 7 + l] = sqrtf((2.0f * l + 1.0f) * 0.07957747155f) * P;
    Pm2 = Pm1; Pm1 = P;
  }
}

__global__ void basis_kernel(const float* __restrict__ sph, const float* __restrict__ proj,
                             const int* __restrict__ ca, float* __restrict__ basis) {
  int i = blockIdx.x * blockDim.x + threadIdx.x;  // T*16
  if (i >= NTRIP * 16) return;
  int t = i >> 4, k = i & 15;
  const float* pr = proj + (size_t)ca[t] * 112 + k;
  const float* sp = sph + (size_t)t * 7;
  float a = 0.0f;
  #pragma unroll
  for (int s = 0; s < 7; ++s) a += sp[s] * pr[s * 16];
  basis[i] = a;
}

__global__ void build_hin_kernel(const float* __restrict__ aemb, const float* __restrict__ z,
                                 const int* __restrict__ at, const int* __restrict__ bt,
                                 u16* __restrict__ o) {
  int i = blockIdx.x * blockDim.x + threadIdx.x;  // N*768
  if (i >= NATOM * 768) return;
  int n = i / 768, c = i % 768;
  float v = (c < 512) ? aemb[(size_t)at[n] * 512 + c]
                      : z[(size_t)bt[n] * 256 + (c - 512)];
  o[i] = f2bf(v);
}

__global__ void build_edgein_kernel(const float* __restrict__ h, const float* __restrict__ rbf,
                                    const int* __restrict__ is, const int* __restrict__ it,
                                    u16* __restrict__ o) {
  long long i = (long long)blockIdx.x * blockDim.x + threadIdx.x;  // E*1152
  if (i >= (long long)NEDGE * 1152) return;
  int e = (int)(i / 1152), c = (int)(i % 1152);
  float v;
  if (c < 512)       v = h[(size_t)is[e] * 512 + c];
  else if (c < 1024) v = h[(size_t)it[e] * 512 + (c - 512)];
  else               v = rbf[(size_t)e * 128 + (c - 1024)];
  o[i] = f2bf(v);
}

__global__ void build_catin_kernel(const float* __restrict__ h, const float* __restrict__ x,
                                   const int* __restrict__ is, const int* __restrict__ it,
                                   u16* __restrict__ o) {
  long long i = (long long)blockIdx.x * blockDim.x + threadIdx.x;  // E*1536
  if (i >= (long long)NEDGE * 1536) return;
  int e = (int)(i / 1536), c = (int)(i % 1536);
  float v;
  if (c < 512)       v = h[(size_t)is[e] * 512 + c];
  else if (c < 1024) v = h[(size_t)it[e] * 512 + (c - 512)];
  else               v = x[(size_t)e * 512 + (c - 1024)];
  o[i] = f2bf(v);
}

__global__ void scatter_mul_kernel(const float* __restrict__ x, const float* __restrict__ g,
                                   const int* __restrict__ idx, float* __restrict__ o) {
  long long i = (long long)blockIdx.x * blockDim.x + threadIdx.x;  // E*512
  if (i >= (long long)NEDGE * 512) return;
  int e = (int)(i >> 9), c = (int)(i & 511);
  atomicAdd(o + (size_t)idx[e] * 512 + c, x[i] * g[i]);
}

__global__ void trip_bil_kernel(const float* __restrict__ basis, const float* __restrict__ P,
                                const int* __restrict__ ba, const int* __restrict__ ca,
                                float* __restrict__ m3) {
  long long i = (long long)blockIdx.x * blockDim.x + threadIdx.x;  // T*64
  if (i >= (long long)NTRIP * 64) return;
  int t = (int)(i >> 6), n = (int)(i & 63);
  const float* pp = P + (size_t)ba[t] * 1024 + n;
  const float* bb = basis + (size_t)t * 16;
  float a = 0.0f;
  #pragma unroll
  for (int k = 0; k < 16; ++k) a += bb[k] * pp[k * 64];
  atomicAdd(m3 + (size_t)ca[t] * 64 + n, a);
}

__global__ void addscale_kernel(const float* __restrict__ a, const float* __restrict__ b,
                                const int* __restrict__ bidx, float s,
                                float* __restrict__ oF, u16* __restrict__ oB,
                                int rows, int C) {
  long long i = (long long)blockIdx.x * blockDim.x + threadIdx.x;
  if (i >= (long long)rows * C) return;
  int r = (int)(i / C), c = (int)(i % C);
  float bv = b[(size_t)(bidx ? bidx[r] : r) * C + c];
  float v = (a[i] + bv) * s;
  if (oF) oF[i] = v;
  if (oB) oB[i] = f2bf(v);
}

__global__ void mul_kernel(const float* __restrict__ a, const float* __restrict__ b,
                           u16* __restrict__ oB, long long n) {
  long long i = (long long)blockIdx.x * blockDim.x + threadIdx.x;
  if (i < n) oB[i] = f2bf(a[i] * b[i]);
}

__global__ void final_E_kernel(const float* __restrict__ Eacc, float* __restrict__ out) {
  int n = blockIdx.x * blockDim.x + threadIdx.x;
  if (n < NATOM) out[n * 4] = Eacc[n];
}

__global__ void final_F_kernel(const float* __restrict__ Facc, const float* __restrict__ V,
                               const int* __restrict__ it, float* __restrict__ out) {
  int e = blockIdx.x * blockDim.x + threadIdx.x;
  if (e >= NEDGE) return;
  float f = Facc[e];
  int a = it[e];
  atomicAdd(out + a * 4 + 1, f * V[e * 3 + 0]);
  atomicAdd(out + a * 4 + 2, f * V[e * 3 + 1]);
  atomicAdd(out + a * 4 + 3, f * V[e * 3 + 2]);
}

// ---------------------------------------------------------------------------
// Host orchestration
// ---------------------------------------------------------------------------
extern "C" void kernel_launch(void* const* d_in, const int* in_sizes, int n_in,
                              void* d_out, int out_size, void* d_ws, size_t ws_size,
                              hipStream_t stream) {
  const float* z_in      = (const float*)d_in[0];
  const float* D_st      = (const float*)d_in[1];
  const float* V_st      = (const float*)d_in[2];
  const float* atom_emb  = (const float*)d_in[3];
  const float* b_atom    = (const float*)d_in[5];
  const int*   atom_types= (const int*)d_in[33];
  const int*   batch     = (const int*)d_in[34];
  const int*   idx_s     = (const int*)d_in[35];
  const int*   idx_t     = (const int*)d_in[36];
  const int*   id3_ba    = (const int*)d_in[37];
  const int*   id3_ca    = (const int*)d_in[38];
  const int*   id_swap   = (const int*)d_in[39];

  const int E = NEDGE, NA = NATOM;

  // ---- workspace bump allocator ----
  char* base = (char*)d_ws;
  size_t off = 0;
  auto alloc  = [&](size_t bytes) -> char* {
    char* p = base + off;
    off = (off + bytes + 255) & ~(size_t)255;
    return p;
  };
  auto allocF = [&](size_t n) { return (float*)alloc(n * 4); };
  auto allocB = [&](size_t n) { return (u16*)alloc(n * 2); };

  struct CvtJob { const float* src; u16* dst; long long n; };
  CvtJob jobs[32]; int njobs = 0;
  auto cvw = [&](int idx, size_t n) -> u16* {
    u16* d = allocB(n);
    jobs[njobs++] = { (const float*)d_in[idx], d, (long long)n };
    return d;
  };

  // bf16 weights
  u16* wb_atom   = cvw(4,  768 * 512);
  u16* wb_edge   = cvw(6,  1152 * 512);
  u16* wb_rbf3   = cvw(7,  128 * 16);
  u16* wb_rbfh   = cvw(9,  128 * 16);
  u16* wb_rbfout = cvw(10, 128 * 16);
  u16* wb_dca    = cvw(11, 3 * 512 * 512);
  u16* wb_dba    = cvw(12, 3 * 512 * 512);
  u16* wb_trip   = cvw(13, 3 * 16 * 512);
  u16* wb_down   = cvw(14, 3 * 512 * 64);
  u16* wb_bil    = cvw(15, 3 * 16 * 64 * 64);
  u16* wb_upca   = cvw(16, 3 * 64 * 512);
  u16* wb_upac   = cvw(17, 3 * 64 * 512);
  u16* wb_resb   = cvw(18, 3 * 2 * 512 * 512);
  u16* wb_resa   = cvw(19, 3 * 4 * 512 * 512);
  u16* wb_aurbf  = cvw(20, 3 * 16 * 512);
  u16* wb_au1    = cvw(21, 3 * 512 * 512);
  u16* wb_aures  = cvw(22, 3 * 6 * 512 * 512);
  u16* wb_concat = cvw(23, 3 * 1536 * 512);
  u16* wb_resm   = cvw(24, 3 * 2 * 512 * 512);
  u16* wb_orbfE  = cvw(25, 4 * 16 * 512);
  u16* wb_oE1    = cvw(26, 4 * 512 * 512);
  u16* wb_oEres  = cvw(27, 4 * 6 * 512 * 512);
  u16* wb_oEfin  = cvw(28, 4 * 512);
  u16* wb_oF1    = cvw(29, 4 * 512 * 512);
  u16* wb_oFres  = cvw(30, 4 * 6 * 512 * 512);
  u16* wb_orbfF  = cvw(31, 4 * 16 * 512);
  u16* wb_oFfin  = cvw(32, 4 * 512);
  u16* wb_cbf3   = allocB(128 * 112);   // repacked from W_cbf3 [7,128,16]

  // activations
  float* rbf_f   = allocF((size_t)E * 128);
  u16*   rbf_b   = allocB((size_t)E * 128);
  u16*   rbf3_b  = allocB((size_t)E * 16);
  u16*   rbfh_b  = allocB((size_t)E * 16);
  u16*   rbfo_b  = allocB((size_t)E * 16);
  float* proj_f  = allocF((size_t)E * 112);
  float* sph_f   = allocF((size_t)NTRIP * 7);
  float* basis_f = allocF((size_t)NTRIP * 16);
  float* h_f     = allocF((size_t)NA * 512);
  u16*   hin_b   = allocB((size_t)NA * 768);
  u16*   cin_b   = allocB((size_t)E * 1536);
  float* m_f     = allocF((size_t)E * 512);
  u16*   m_b     = allocB((size_t)E * 512);
  float* x_f     = allocF((size_t)E * 512);
  u16*   x_b     = allocB((size_t)E * 512);
  float* t1_f    = allocF((size_t)E * 512);
  u16*   t1_b    = allocB((size_t)E * 512);
  float* t2_f    = allocF((size_t)E * 512);
  u16*   t2_b    = allocB((size_t)E * 512);
  float* g1_f    = allocF((size_t)E * 512);
  float* P_f     = allocF((size_t)E * 1024);
  float* m3_f    = allocF((size_t)E * 64);
  u16*   m3_b    = allocB((size_t)E * 64);
  u16*   dn_b    = allocB((size_t)E * 64);
  float* xa_f    = allocF((size_t)NA * 512);
  u16*   xa_b    = allocB((size_t)NA * 512);
  float* aa_f    = allocF((size_t)NA * 512);
  u16*   aa_b    = allocB((size_t)NA * 512);
  u16*   tb_b    = allocB((size_t)NA * 512);
  float* Eacc    = allocF((size_t)NA);
  float* Facc    = allocF((size_t)E);

  if (off > ws_size) return;  // workspace insufficient; cannot report

  // ---- launch helpers ----
  auto gemm = [&](const u16* A, int lda, const u16* B, int ldb,
                  int M, int N, int K,
                  const float* bias, const float* mulp, const float* addp, float asc,
                  float* oF, u16* oB, int ldc, int flags) {
    dim3 grid((unsigned)((N + 127) / 128), (unsigned)((M + 127) / 128));
    wmma_gemm_kernel<<<grid, 256, 0, stream>>>(A, lda, B, ldb, M, N, K,
                                               bias, mulp, addp, asc, oF, oB, ldc, flags);
  };
  auto cvt = [&](const float* s, u16* d, long long n) {
    cvt_kernel<<<dim3((unsigned)((n + 255) / 256)), 256, 0, stream>>>(s, d, n);
  };
  auto addscale = [&](const float* a, const float* b, const int* bidx, float s,
                      float* oF, u16* oB, int rows, int C) {
    long long n = (long long)rows * C;
    addscale_kernel<<<dim3((unsigned)((n + 255) / 256)), 256, 0, stream>>>(a, b, bidx, s, oF, oB, rows, C);
  };
  // residual stack: x = (x + silu(silu(x@W1)@W2)) / sqrt(2), nlayers times
  auto res_stack = [&](const u16* Wbase, int nlayers, float* xf, u16* xbf, u16* tmpbf, int M) {
    for (int l = 0; l < nlayers; ++l) {
      const u16* W1 = Wbase + (size_t)l * 2 * 512 * 512;
      const u16* W2 = W1 + 512 * 512;
      gemm(xbf, 512, W1, 512, M, 512, 512, 0, 0, 0, 1.0f, 0, tmpbf, 512, FLAG_SILU);
      gemm(tmpbf, 512, W2, 512, M, 512, 512, 0, 0, xf, INV_SQRT2F, xf, xbf, 512, FLAG_SILU);
    }
  };

  // ---- weight conversions ----
  for (int j = 0; j < njobs; ++j) {
    long long n = jobs[j].n;
    cvt_kernel<<<dim3((unsigned)((n + 255) / 256)), 256, 0, stream>>>(jobs[j].src, jobs[j].dst, n);
  }
  pack_cbf3_kernel<<<dim3((7 * 128 * 16 + 255) / 256), 256, 0, stream>>>((const float*)d_in[8], wb_cbf3);

  // ---- basis precompute ----
  rbf_kernel<<<dim3((E * 128 + 255) / 256), 256, 0, stream>>>(D_st, rbf_f, rbf_b);
  gemm(rbf_b, 128, wb_rbf3,   16,  E, 16,  128, 0, 0, 0, 1.0f, 0, rbf3_b, 16,  0);
  gemm(rbf_b, 128, wb_rbfh,   16,  E, 16,  128, 0, 0, 0, 1.0f, 0, rbfh_b, 16,  0);
  gemm(rbf_b, 128, wb_rbfout, 16,  E, 16,  128, 0, 0, 0, 1.0f, 0, rbfo_b, 16,  0);
  gemm(rbf_b, 128, wb_cbf3,   112, E, 112, 128, 0, 0, 0, 1.0f, proj_f, 0, 112, 0);
  sph_kernel<<<dim3((NTRIP + 255) / 256), 256, 0, stream>>>(V_st, id3_ca, id3_ba, sph_f);
  basis_kernel<<<dim3((unsigned)(((long long)NTRIP * 16 + 255) / 256)), 256, 0, stream>>>(sph_f, proj_f, id3_ca, basis_f);

  // ---- atom + edge embeddings ----
  build_hin_kernel<<<dim3((NA * 768 + 255) / 256), 256, 0, stream>>>(atom_emb, z_in, atom_types, batch, hin_b);
  gemm(hin_b, 768, wb_atom, 512, NA, 512, 768, b_atom, 0, 0, 1.0f, h_f, 0, 512, 0);
  build_edgein_kernel<<<dim3((unsigned)(((long long)E * 1152 + 255) / 256)), 256, 0, stream>>>(h_f, rbf_f, idx_s, idx_t, cin_b);
  gemm(cin_b, 1152, wb_edge, 512, E, 512, 1152, 0, 0, 0, 1.0f, m_f, m_b, 512, FLAG_SILU);

  hipMemsetAsync(Eacc, 0, (size_t)NA * 4, stream);
  hipMemsetAsync(Facc, 0, (size_t)E * 4, stream);

  // ---- output block ----
  auto out_block = [&](int o) {
    size_t o512 = (size_t)o * 512 * 512;
    // xE path
    gemm(rbfo_b, 16, wb_orbfE + (size_t)o * 16 * 512, 512, E, 512, 16, 0, 0, 0, 1.0f, g1_f, 0, 512, 0);
    hipMemsetAsync(xa_f, 0, (size_t)NA * 512 * 4, stream);
    scatter_mul_kernel<<<dim3((unsigned)(((long long)E * 512 + 255) / 256)), 256, 0, stream>>>(m_f, g1_f, idx_t, xa_f);
    cvt(xa_f, xa_b, (long long)NA * 512);
    gemm(xa_b, 512, wb_oE1 + o512, 512, NA, 512, 512, 0, 0, 0, 1.0f, aa_f, aa_b, 512, FLAG_SILU);
    res_stack(wb_oEres + (size_t)o * 6 * 512 * 512, 3, aa_f, aa_b, tb_b, NA);
    gemm(aa_b, 512, wb_oEfin + (size_t)o * 512, 1, NA, 1, 512, 0, 0, Eacc, 1.0f, Eacc, 0, 1, 0);
    // xF path
    gemm(m_b, 512, wb_oF1 + o512, 512, E, 512, 512, 0, 0, 0, 1.0f, t1_f, t1_b, 512, FLAG_SILU);
    res_stack(wb_oFres + (size_t)o * 6 * 512 * 512, 3, t1_f, t1_b, t2_b, E);
    gemm(rbfo_b, 16, wb_orbfF + (size_t)o * 16 * 512, 512, E, 512, 16, 0, 0, 0, 1.0f, g1_f, 0, 512, 0);
    mul_kernel<<<dim3((unsigned)(((long long)E * 512 + 255) / 256)), 256, 0, stream>>>(t1_f, g1_f, t2_b, (long long)E * 512);
    gemm(t2_b, 512, wb_oFfin + (size_t)o * 512, 1, E, 1, 512, 0, 0, Facc, 1.0f, Facc, 0, 1, 0);
  };

  out_block(0);

  // ---- interaction blocks ----
  for (int i = 0; i < 3; ++i) {
    size_t i512 = (size_t)i * 512 * 512;
    // triplet interaction
    gemm(rbf3_b, 16, wb_trip + (size_t)i * 16 * 512, 512, E, 512, 16, 0, 0, 0, 1.0f, g1_f, 0, 512, 0);
    gemm(m_b, 512, wb_dba + i512, 512, E, 512, 512, 0, g1_f, 0, 1.0f, 0, t1_b, 512, FLAG_SILU);
    gemm(t1_b, 512, wb_down + (size_t)i * 512 * 64, 64, E, 64, 512, 0, 0, 0, 1.0f, 0, dn_b, 64, FLAG_SILU);
    for (int k = 0; k < 16; ++k)
      gemm(dn_b, 64, wb_bil + (size_t)i * 65536 + (size_t)k * 4096, 64,
           E, 64, 64, 0, 0, 0, 1.0f, P_f + (size_t)k * 64, 0, 1024, 0);
    hipMemsetAsync(m3_f, 0, (size_t)E * 64 * 4, stream);
    trip_bil_kernel<<<dim3((unsigned)(((long long)NTRIP * 64 + 255) / 256)), 256, 0, stream>>>(basis_f, P_f, id3_ba, id3_ca, m3_f);
    cvt(m3_f, m3_b, (long long)E * 64);
    gemm(m3_b, 64, wb_upca + (size_t)i * 64 * 512, 512, E, 512, 64, 0, 0, 0, 1.0f, t1_f, 0, 512, FLAG_SILU);
    gemm(m3_b, 64, wb_upac + (size_t)i * 64 * 512, 512, E, 512, 64, 0, 0, 0, 1.0f, t2_f, 0, 512, FLAG_SILU);
    addscale(t1_f, t2_f, id_swap, INV_SQRT2F, t1_f, 0, E, 512);            // x3
    gemm(m_b, 512, wb_dca + i512, 512, E, 512, 512, 0, 0, t1_f, INV_SQRT2F, x_f, x_b, 512, FLAG_SILU);
    res_stack(wb_resb + (size_t)i * 2 * 512 * 512, 1, x_f, x_b, t1_b, E);
    addscale(x_f, m_f, 0, INV_SQRT2F, x_f, x_b, E, 512);                   // + m_skip
    res_stack(wb_resa + (size_t)i * 4 * 512 * 512, 2, x_f, x_b, t1_b, E);
    // atom update
    gemm(rbfh_b, 16, wb_aurbf + (size_t)i * 16 * 512, 512, E, 512, 16, 0, 0, 0, 1.0f, g1_f, 0, 512, 0);
    hipMemsetAsync(xa_f, 0, (size_t)NA * 512 * 4, stream);
    scatter_mul_kernel<<<dim3((unsigned)(((long long)E * 512 + 255) / 256)), 256, 0, stream>>>(x_f, g1_f, idx_t, xa_f);
    cvt(xa_f, xa_b, (long long)NA * 512);
    gemm(xa_b, 512, wb_au1 + i512, 512, NA, 512, 512, 0, 0, 0, 1.0f, aa_f, aa_b, 512, FLAG_SILU);
    res_stack(wb_aures + (size_t)i * 6 * 512 * 512, 3, aa_f, aa_b, tb_b, NA);
    addscale(h_f, aa_f, 0, INV_SQRT2F, h_f, 0, NA, 512);                   // h update
    // concat rebuild of edge embedding
    build_catin_kernel<<<dim3((unsigned)(((long long)E * 1536 + 255) / 256)), 256, 0, stream>>>(h_f, x_f, idx_s, idx_t, cin_b);
    gemm(cin_b, 1536, wb_concat + (size_t)i * 1536 * 512, 512, E, 512, 1536, 0, 0, 0, 1.0f, t1_f, t1_b, 512, FLAG_SILU);
    res_stack(wb_resm + (size_t)i * 2 * 512 * 512, 1, t1_f, t1_b, t2_b, E);
    addscale(t1_f, m_f, 0, INV_SQRT2F, m_f, m_b, E, 512);                  // m update

    out_block(i + 1);
  }

  // ---- final assembly: out[N,4] = [E_at, F_at(3)] ----
  hipMemsetAsync(d_out, 0, (size_t)NA * 4 * sizeof(float), stream);
  final_E_kernel<<<dim3((NA + 255) / 256), 256, 0, stream>>>(Eacc, (float*)d_out);
  final_F_kernel<<<dim3((E + 255) / 256), 256, 0, stream>>>(Facc, V_st, idx_t, (float*)d_out);
}
